// MambaBlocks_9148280340644
// MI455X (gfx1250) — matrix operand order repaired and checked by reading, not compiled, not run on hardware
//
#include <hip/hip_runtime.h>
#include <hip/hip_bf16.h>
#include <math.h>
#include <stdint.h>

// ---------- model dims ----------
#define B_     8
#define L_     1024
#define D_     768
#define DEPTH_ 12
#define DIN_   1536      // 2*D
#define N_     16
#define DTR_   48
#define KC_    4         // conv kernel
#define NCLS_  22
#define ROWS_  (B_*L_)   // 8192
#define EPS_   1e-5f
#define XDW_   (DTR_ + 2*N_)   // 80
#define DTK_   64              // dt GEMM K padded 48 -> 64

typedef __attribute__((ext_vector_type(16))) __bf16 v16bf;
typedef __attribute__((ext_vector_type(8)))  __bf16 bf16x8;
typedef __attribute__((ext_vector_type(8)))  float  v8f;

__device__ __forceinline__ float siluf(float x) { return x / (1.f + __expf(-x)); }

// =====================================================================
// Aligned bf16 GEMM, CDNA5 async global->LDS, triple-buffered,
// one barrier per 64-wide K-tile (16 WMMAs / barrier).
// Requires: M % 128 == 0, N % 128 == 0, K % 64 == 0.
// C(MxN) = A(MxK) * W(NxK)^T, fp32 accumulate.
// 256 threads = 8 waves (2m x 4n); wave computes 64x32 per K-step.
// =====================================================================
#define BM2 128
#define BN2 128
#define BK2 64

__global__ __launch_bounds__(256) void gemm_bf16_nt_async(
    const __bf16* __restrict__ A, const __bf16* __restrict__ W,
    float* __restrict__ C, int M, int N, int K)
{
    __shared__ __attribute__((aligned(16))) __bf16 sA[3][BM2][BK2];
    __shared__ __attribute__((aligned(16))) __bf16 sB[3][BN2][BK2];

    const int tid  = threadIdx.x;
    const int wid  = tid >> 5;
    const int lane = tid & 31;
    const int m0   = blockIdx.y * BM2;
    const int n0   = blockIdx.x * BN2;
    const int wmB  = (wid & 1) * 64;   // wave m-offset (2 wave rows)
    const int wnB  = (wid >> 1) * 32;  // wave n-offset (4 wave cols)

    v8f acc[4][2];
#pragma unroll
    for (int mt = 0; mt < 4; ++mt)
#pragma unroll
        for (int nt = 0; nt < 2; ++nt)
            acc[mt][nt] = (v8f){};

    // staging: thread -> (row = tid/2, col32 = (tid&1)*32); 32 elems = 4 x b128
    const int sr = tid >> 1;          // 0..127
    const int sc = (tid & 1) * 32;    // 0 or 32

    const __bf16* gA = A + (size_t)(m0 + sr) * K + sc;
    const __bf16* gB = W + (size_t)(n0 + sr) * K + sc;

    // LDS byte addresses (generic LDS pointer truncates to LDS offset)
    uint32_t ldsA[3], ldsB[3];
#pragma unroll
    for (int p = 0; p < 3; ++p) {
        ldsA[p] = (uint32_t)(uintptr_t)&sA[p][sr][sc];
        ldsB[p] = (uint32_t)(uintptr_t)&sB[p][sr][sc];
    }

#define ISSUE_ONE(dst, src, off)                                                 \
    asm volatile("global_load_async_to_lds_b128 %0, %1, off offset:" #off        \
                 :: "v"(dst), "v"(src) : "memory")

    // 8 async B128 loads per tile per thread; offset applies to LDS + global
#define ISSUE_TILE(k0, p)                                                        \
    do {                                                                         \
        ISSUE_ONE(ldsA[p], gA + (k0), 0);                                        \
        ISSUE_ONE(ldsA[p], gA + (k0), 16);                                       \
        ISSUE_ONE(ldsA[p], gA + (k0), 32);                                       \
        ISSUE_ONE(ldsA[p], gA + (k0), 48);                                       \
        ISSUE_ONE(ldsB[p], gB + (k0), 0);                                        \
        ISSUE_ONE(ldsB[p], gB + (k0), 16);                                       \
        ISSUE_ONE(ldsB[p], gB + (k0), 32);                                       \
        ISSUE_ONE(ldsB[p], gB + (k0), 48);                                       \
    } while (0)

    const int nk = K / BK2;
    const int mrow  = lane & 15;
    const int abase = (lane >> 4) * 8;   // A: half-K offset
    const int kb    = (lane >> 4) * 8;   // B: half-K offset

    auto compute = [&](int p) {
#pragma unroll
        for (int kk = 0; kk < 2; ++kk) {          // two 32-wide WMMA K-steps
            const int ko = kk * 32;
            v16bf af[4], bfr[2];
#pragma unroll
            for (int mt = 0; mt < 4; ++mt) {
                const int r = wmB + mt * 16 + mrow;
                bf16x8 lo = *(const bf16x8*)&sA[p][r][ko + abase];
                bf16x8 hi = *(const bf16x8*)&sA[p][r][ko + 16 + abase];
#pragma unroll
                for (int j = 0; j < 8; ++j) { af[mt][j] = lo[j]; af[mt][j + 8] = hi[j]; }
            }
#pragma unroll
            for (int nt = 0; nt < 2; ++nt) {
                const int r = wnB + nt * 16 + mrow;
                bf16x8 lo = *(const bf16x8*)&sB[p][r][ko + kb];
                bf16x8 hi = *(const bf16x8*)&sB[p][r][ko + 16 + kb];
#pragma unroll
                for (int j = 0; j < 8; ++j) { bfr[nt][j] = lo[j]; bfr[nt][j + 8] = hi[j]; }
            }
#pragma unroll
            for (int mt = 0; mt < 4; ++mt)
#pragma unroll
                for (int nt = 0; nt < 2; ++nt)
                    acc[mt][nt] = __builtin_amdgcn_wmma_f32_16x16x32_bf16(
                        false, af[mt], false, bfr[nt], (short)0, acc[mt][nt], false, false);
        }
    };

    // steady-state step with compile-time buffer phase q
#define STEP_STEADY(q, ktv)                                                      \
    do {                                                                         \
        asm volatile("s_wait_asynccnt 0x8" ::: "memory");                        \
        __syncthreads();                                                         \
        compute(q);                                                              \
        ISSUE_TILE(((ktv) + 2) * BK2, ((q) + 2) % 3);                            \
    } while (0)

    // prologue: fill pipeline
    ISSUE_TILE(0, 0);
    if (nk > 1) ISSUE_TILE(BK2, 1);

    int kt = 0;
    // unrolled-by-3 steady state: constant buffer indices, no %3 in hot loop
    for (; kt + 5 <= nk; kt += 3) {
        STEP_STEADY(0, kt);
        STEP_STEADY(1, kt + 1);
        STEP_STEADY(2, kt + 2);
    }
    // leftover steady iterations (at most 2)
    for (; kt + 2 < nk; ++kt) {
        asm volatile("s_wait_asynccnt 0x8" ::: "memory");
        __syncthreads();
        compute(kt % 3);
        ISSUE_TILE((kt + 2) * BK2, (kt + 2) % 3);
    }
    // penultimate tile (exists when nk >= 2)
    if (kt < nk - 1) {
        asm volatile("s_wait_asynccnt 0x8" ::: "memory");
        __syncthreads();
        compute(kt % 3);
        ++kt;
    }
    // last tile
    asm volatile("s_wait_asynccnt 0x0" ::: "memory");
    __syncthreads();
    compute(kt % 3);
#undef STEP_STEADY
#undef ISSUE_TILE
#undef ISSUE_ONE

    // ---- store C (16x16 f32 layout: row = v + 8*(lane>=16), col = lane&15) ----
    const int cn     = lane & 15;
    const int rshift = (lane >> 4) * 8;
#pragma unroll
    for (int mt = 0; mt < 4; ++mt) {
        const int rbase = m0 + wmB + mt * 16 + rshift;
#pragma unroll
        for (int nt = 0; nt < 2; ++nt) {
            const int cidx = n0 + wnB + nt * 16 + cn;
#pragma unroll
            for (int v = 0; v < 8; ++v)
                C[(size_t)(rbase + v) * N + cidx] = acc[mt][nt][v];
        }
    }
}

// =====================================================================
// Generic guarded bf16 GEMM (only used for x_proj, N = 80)
// =====================================================================
#define BM 64
#define BN 128
#define BK 32

__global__ __launch_bounds__(256) void gemm_bf16_nt(
    const __bf16* __restrict__ A, const __bf16* __restrict__ W,
    float* __restrict__ C, int M, int N, int K)
{
    __shared__ __attribute__((aligned(16))) __bf16 sA[BM][BK];
    __shared__ __attribute__((aligned(16))) __bf16 sB[BN][BK];

    const int tid  = threadIdx.x;
    const int wid  = tid >> 5;
    const int lane = tid & 31;
    const int m0   = blockIdx.y * BM;
    const int n0   = blockIdx.x * BN;
    const int wm   = (wid & 1) * 32;
    const int wn   = (wid >> 1) * 32;

    v8f acc00 = {}, acc01 = {}, acc10 = {}, acc11 = {};

    const int ar = tid >> 2;
    const int ac = (tid & 3) * 8;

    for (int k0 = 0; k0 < K; k0 += BK) {
        {
            int row = m0 + ar;
            bf16x8 v = {};
            if (row < M) v = *(const bf16x8*)(A + (size_t)row * K + k0 + ac);
            *(bf16x8*)&sA[ar][ac] = v;
        }
#pragma unroll
        for (int i = 0; i < 2; ++i) {
            int idx = tid + i * 256;
            int r = idx >> 2, c = (idx & 3) * 8;
            int row = n0 + r;
            bf16x8 v = {};
            if (row < N) v = *(const bf16x8*)(W + (size_t)row * K + k0 + c);
            *(bf16x8*)&sB[r][c] = v;
        }
        __syncthreads();

        const int mrow  = lane & 15;
        const int abase = (lane >> 4) * 8;
        const int kb    = (lane >> 4) * 8;
        v16bf a0, a1, b0, b1;
        {
            bf16x8 lo = *(const bf16x8*)&sA[wm + mrow][abase];
            bf16x8 hi = *(const bf16x8*)&sA[wm + mrow][16 + abase];
#pragma unroll
            for (int j = 0; j < 8; ++j) { a0[j] = lo[j]; a0[j + 8] = hi[j]; }
            lo = *(const bf16x8*)&sA[wm + 16 + mrow][abase];
            hi = *(const bf16x8*)&sA[wm + 16 + mrow][16 + abase];
#pragma unroll
            for (int j = 0; j < 8; ++j) { a1[j] = lo[j]; a1[j + 8] = hi[j]; }
            lo = *(const bf16x8*)&sB[wn + mrow][kb];
            hi = *(const bf16x8*)&sB[wn + mrow][16 + kb];
#pragma unroll
            for (int j = 0; j < 8; ++j) { b0[j] = lo[j]; b0[j + 8] = hi[j]; }
            lo = *(const bf16x8*)&sB[wn + 16 + mrow][kb];
            hi = *(const bf16x8*)&sB[wn + 16 + mrow][16 + kb];
#pragma unroll
            for (int j = 0; j < 8; ++j) { b1[j] = lo[j]; b1[j + 8] = hi[j]; }
        }

        acc00 = __builtin_amdgcn_wmma_f32_16x16x32_bf16(false, a0, false, b0, (short)0, acc00, false, false);
        acc01 = __builtin_amdgcn_wmma_f32_16x16x32_bf16(false, a0, false, b1, (short)0, acc01, false, false);
        acc10 = __builtin_amdgcn_wmma_f32_16x16x32_bf16(false, a1, false, b0, (short)0, acc10, false, false);
        acc11 = __builtin_amdgcn_wmma_f32_16x16x32_bf16(false, a1, false, b1, (short)0, acc11, false, false);
        __syncthreads();
    }

    const int cn     = lane & 15;
    const int rshift = (lane >> 4) * 8;
#pragma unroll
    for (int v = 0; v < 8; ++v) {
        int r0 = m0 + wm + rshift + v;
        int r1 = r0 + 16;
        int c0 = n0 + wn + cn;
        int c1 = c0 + 16;
        if (r0 < M) {
            if (c0 < N) C[(size_t)r0 * N + c0] = acc00[v];
            if (c1 < N) C[(size_t)r0 * N + c1] = acc01[v];
        }
        if (r1 < M) {
            if (c0 < N) C[(size_t)r1 * N + c0] = acc10[v];
            if (c1 < N) C[(size_t)r1 * N + c1] = acc11[v];
        }
    }
}

// =====================================================================
// Elementwise / helper kernels
// =====================================================================
__global__ void embed_kernel(const int* __restrict__ x, const float* __restrict__ emb,
                             const float* __restrict__ pos, float* __restrict__ h)
{
    size_t i = (size_t)blockIdx.x * blockDim.x + threadIdx.x;
    size_t total = (size_t)ROWS_ * D_;
    if (i >= total) return;
    int j   = (int)(i % D_);
    int row = (int)(i / D_);
    int l   = row % L_;
    int tok = x[row];
    h[i] = emb[(size_t)tok * D_ + j] + pos[(size_t)l * D_ + j];
}

__global__ void cvt_bf16_kernel(const float* __restrict__ s, __bf16* __restrict__ d, size_t n)
{
    size_t stride = (size_t)gridDim.x * blockDim.x;
    for (size_t i = (size_t)blockIdx.x * blockDim.x + threadIdx.x; i < n; i += stride)
        d[i] = (__bf16)s[i];
}

__global__ void pack_dtw_kernel(const float* __restrict__ s, __bf16* __restrict__ d)
{
    size_t i = (size_t)blockIdx.x * blockDim.x + threadIdx.x;
    size_t total = (size_t)DEPTH_ * DIN_ * DTK_;
    if (i >= total) return;
    int col = (int)(i % DTK_);
    size_t row = i / DTK_;
    d[i] = (__bf16)((col < DTR_) ? s[row * DTR_ + col] : 0.f);
}

__global__ __launch_bounds__(256) void prenorm_kernel(
    const float* __restrict__ h, float* __restrict__ resid,
    const float* __restrict__ w, __bf16* __restrict__ out, int first)
{
    __shared__ float red[256];
    int row = blockIdx.x;
    const float* hr = h + (size_t)row * D_;
    float* rr = resid + (size_t)row * D_;
    float vals[3];
    float ss = 0.f;
#pragma unroll
    for (int i = 0; i < 3; ++i) {
        int j = threadIdx.x + i * 256;
        float v = hr[j];
        if (!first) v += rr[j];
        rr[j] = v;
        vals[i] = v;
        ss += v * v;
    }
    red[threadIdx.x] = ss;
    __syncthreads();
    for (int s = 128; s > 0; s >>= 1) {
        if (threadIdx.x < s) red[threadIdx.x] += red[threadIdx.x + s];
        __syncthreads();
    }
    float scale = rsqrtf(red[0] / (float)D_ + EPS_);
#pragma unroll
    for (int i = 0; i < 3; ++i) {
        int j = threadIdx.x + i * 256;
        out[(size_t)row * D_ + j] = (__bf16)(vals[i] * scale * w[j]);
    }
}

__global__ void conv_silu_kernel(const float* __restrict__ xz,
                                 const float* __restrict__ cw, const float* __restrict__ cb,
                                 float* __restrict__ uf, __bf16* __restrict__ ub)
{
    size_t i = (size_t)blockIdx.x * blockDim.x + threadIdx.x;
    size_t total = (size_t)ROWS_ * DIN_;
    if (i >= total) return;
    int c   = (int)(i % DIN_);
    int row = (int)(i / DIN_);
    int l   = row % L_;
    float acc = cb[c];
#pragma unroll
    for (int k = 0; k < KC_; ++k) {
        int lk = l - (KC_ - 1) + k;
        if (lk >= 0)
            acc += xz[(size_t)(row - (KC_ - 1) + k) * (2 * DIN_) + c] * cw[c * KC_ + k];
    }
    float v = siluf(acc);
    uf[i] = v;
    ub[i] = (__bf16)v;
}

__global__ void pack_dtlo_kernel(const float* __restrict__ xdbl, __bf16* __restrict__ out)
{
    size_t i = (size_t)blockIdx.x * blockDim.x + threadIdx.x;
    size_t total = (size_t)ROWS_ * DTK_;
    if (i >= total) return;
    int col = (int)(i % DTK_);
    size_t row = i / DTK_;
    out[i] = (__bf16)((col < DTR_) ? xdbl[row * XDW_ + col] : 0.f);
}

__global__ void softplus_kernel(float* __restrict__ dt, const float* __restrict__ b)
{
    size_t i = (size_t)blockIdx.x * blockDim.x + threadIdx.x;
    size_t total = (size_t)ROWS_ * DIN_;
    if (i >= total) return;
    int c = (int)(i % DIN_);
    float x = dt[i] + b[c];
    dt[i] = (x > 20.f) ? x : log1pf(__expf(x));
}

// selective scan with chunked B/C staging (64 timesteps per LDS refill)
#define TS_ 64
__global__ __launch_bounds__(256) void scan_kernel(
    const float* __restrict__ uf, const float* __restrict__ dt,
    const float* __restrict__ xdbl, const float* __restrict__ xz,
    const float* __restrict__ a_log, const float* __restrict__ d_skip,
    __bf16* __restrict__ yb)
{
    const int blocksPerB = DIN_ / 256;
    int b = blockIdx.x / blocksPerB;
    int c = (blockIdx.x % blocksPerB) * 256 + threadIdx.x;
    __shared__ __attribute__((aligned(16))) float sBC[TS_][32]; // [t][0..15]=B, [16..31]=C
    float A[N_], st[N_];
#pragma unroll
    for (int n = 0; n < N_; ++n) {
        A[n]  = -__expf(a_log[(size_t)c * N_ + n]);
        st[n] = 0.f;
    }
    float dsk = d_skip[c];
    const int lr = threadIdx.x >> 2;        // 0..63 : timestep within chunk
    const int lq = (threadIdx.x & 3) * 8;   // 0,8,16,24 : offset within B|C row

    for (int t0 = 0; t0 < L_; t0 += TS_) {
        __syncthreads();  // previous chunk fully consumed
        {
            const float* src = xdbl + ((size_t)b * L_ + t0 + lr) * XDW_ + DTR_ + lq;
            float4 v0 = *(const float4*)(src);
            float4 v1 = *(const float4*)(src + 4);
            *(float4*)&sBC[lr][lq]     = v0;
            *(float4*)&sBC[lr][lq + 4] = v1;
        }
        __syncthreads();

        for (int tt = 0; tt < TS_; ++tt) {
            size_t row = (size_t)b * L_ + t0 + tt;
            float dtv = dt[row * DIN_ + c];
            float uv  = uf[row * DIN_ + c];
            float du  = dtv * uv;
            float y = 0.f;
#pragma unroll
            for (int n = 0; n < N_; ++n) {
                float dA = __expf(dtv * A[n]);
                st[n] = dA * st[n] + du * sBC[tt][n];
                y += st[n] * sBC[tt][16 + n];
            }
            float z = xz[row * (2 * DIN_) + DIN_ + c];
            yb[row * DIN_ + c] = (__bf16)((y + uv * dsk) * siluf(z));
        }
    }
}

__global__ __launch_bounds__(256) void final_head_kernel(
    const float* __restrict__ h, const float* __restrict__ resid,
    const float* __restrict__ wf, const float* __restrict__ hw,
    const float* __restrict__ hb, float* __restrict__ out)
{
    __shared__ float hn[D_];
    __shared__ float red[256];
    int b = blockIdx.x;
    size_t base = (size_t)b * L_ * D_;
    float vals[3];
    float ss = 0.f;
#pragma unroll
    for (int i = 0; i < 3; ++i) {
        int j = threadIdx.x + i * 256;
        float v = h[base + j] + resid[base + j];
        vals[i] = v;
        ss += v * v;
    }
    red[threadIdx.x] = ss;
    __syncthreads();
    for (int s = 128; s > 0; s >>= 1) {
        if (threadIdx.x < s) red[threadIdx.x] += red[threadIdx.x + s];
        __syncthreads();
    }
    float scale = rsqrtf(red[0] / (float)D_ + EPS_);
#pragma unroll
    for (int i = 0; i < 3; ++i) {
        int j = threadIdx.x + i * 256;
        hn[j] = vals[i] * scale * wf[j];
    }
    __syncthreads();
    if (threadIdx.x < NCLS_) {
        float acc = hb[threadIdx.x];
        for (int j = 0; j < D_; ++j)
            acc += hn[j] * hw[threadIdx.x * D_ + j];
        out[b * NCLS_ + threadIdx.x] = acc;
    }
}

// =====================================================================
// Host orchestration
// =====================================================================
extern "C" void kernel_launch(void* const* d_in, const int* in_sizes, int n_in,
                              void* d_out, int out_size, void* d_ws, size_t ws_size,
                              hipStream_t stream)
{
    (void)in_sizes; (void)n_in; (void)out_size; (void)ws_size;

    const int*   x    = (const int*)  d_in[0];
    const float* emb  = (const float*)d_in[1];
    const float* pos  = (const float*)d_in[2];
    const float* ipw  = (const float*)d_in[3];
    const float* cw   = (const float*)d_in[4];
    const float* cb   = (const float*)d_in[5];
    const float* xpw  = (const float*)d_in[6];
    const float* dtw  = (const float*)d_in[7];
    const float* dtb  = (const float*)d_in[8];
    const float* alog = (const float*)d_in[9];
    const float* dsk  = (const float*)d_in[10];
    const float* opw  = (const float*)d_in[11];
    const float* nw   = (const float*)d_in[12];
    const float* nfw  = (const float*)d_in[13];
    const float* hw   = (const float*)d_in[14];
    const float* hb   = (const float*)d_in[15];
    float* out = (float*)d_out;

    char* ws = (char*)d_ws;
    size_t cur = 0;
    auto alloc = [&](size_t bytes) -> char* {
        char* p = ws + cur;
        cur = (cur + bytes + 255) & ~(size_t)255;
        return p;
    };

    float*  hbuf  = (float*) alloc((size_t)ROWS_ * D_   * 4);
    float*  resid = (float*) alloc((size_t)ROWS_ * D_   * 4);
    __bf16* hnb   = (__bf16*)alloc((size_t)ROWS_ * D_   * 2);
    float*  xz    = (float*) alloc((size_t)ROWS_ * 2 * DIN_ * 4);
    float*  uf    = (float*) alloc((size_t)ROWS_ * DIN_ * 4);
    __bf16* ub    = (__bf16*)alloc((size_t)ROWS_ * DIN_ * 2);
    float*  xdbl  = (float*) alloc((size_t)ROWS_ * XDW_ * 4);
    __bf16* dtlob = (__bf16*)alloc((size_t)ROWS_ * DTK_ * 2);
    float*  dtbuf = (float*) alloc((size_t)ROWS_ * DIN_ * 4);
    __bf16* yb    = (__bf16*)alloc((size_t)ROWS_ * DIN_ * 2);
    __bf16* ipwb  = (__bf16*)alloc((size_t)DEPTH_ * 2 * DIN_ * D_ * 2);
    __bf16* xpwb  = (__bf16*)alloc((size_t)DEPTH_ * XDW_ * DIN_ * 2);
    __bf16* dtwb  = (__bf16*)alloc((size_t)DEPTH_ * DIN_ * DTK_ * 2);
    __bf16* opwb  = (__bf16*)alloc((size_t)DEPTH_ * D_ * DIN_ * 2);

    auto gemm_fast = [&](const __bf16* A, const __bf16* W, float* C, int M, int N, int K) {
        dim3 g(N / BN2, M / BM2);
        gemm_bf16_nt_async<<<g, 256, 0, stream>>>(A, W, C, M, N, K);
    };
    auto gemm_edge = [&](const __bf16* A, const __bf16* W, float* C, int M, int N, int K) {
        dim3 g((N + BN - 1) / BN, (M + BM - 1) / BM);
        gemm_bf16_nt<<<g, 256, 0, stream>>>(A, W, C, M, N, K);
    };

    // ---- one-time weight conversion ----
    {
        size_t n1 = (size_t)DEPTH_ * 2 * DIN_ * D_;
        size_t n2 = (size_t)DEPTH_ * XDW_ * DIN_;
        size_t n3 = (size_t)DEPTH_ * D_ * DIN_;
        cvt_bf16_kernel<<<4096, 256, 0, stream>>>(ipw, ipwb, n1);
        cvt_bf16_kernel<<<1024, 256, 0, stream>>>(xpw, xpwb, n2);
        cvt_bf16_kernel<<<4096, 256, 0, stream>>>(opw, opwb, n3);
        size_t ndt = (size_t)DEPTH_ * DIN_ * DTK_;
        pack_dtw_kernel<<<(int)((ndt + 255) / 256), 256, 0, stream>>>(dtw, dtwb);
    }

    // ---- embedding ----
    {
        size_t total = (size_t)ROWS_ * D_;
        embed_kernel<<<(int)((total + 255) / 256), 256, 0, stream>>>(x, emb, pos, hbuf);
    }

    const size_t totUD = (size_t)ROWS_ * DIN_;
    const int    gUD   = (int)((totUD + 255) / 256);

    // ---- layer loop ----
    for (int i = 0; i < DEPTH_; ++i) {
        prenorm_kernel<<<ROWS_, 256, 0, stream>>>(hbuf, resid, nw + (size_t)i * D_, hnb, i == 0);

        // in_proj: (8192x768) x (3072x768)^T -> xz   [aligned, nk=12]
        gemm_fast(hnb, ipwb + (size_t)i * 2 * DIN_ * D_, xz, ROWS_, 2 * DIN_, D_);

        conv_silu_kernel<<<gUD, 256, 0, stream>>>(xz, cw + (size_t)i * DIN_ * KC_,
                                                  cb + (size_t)i * DIN_, uf, ub);

        // x_proj: (8192x1536) x (80x1536)^T -> xdbl  [edge, N=80]
        gemm_edge(ub, xpwb + (size_t)i * XDW_ * DIN_, xdbl, ROWS_, XDW_, DIN_);

        {
            size_t tot = (size_t)ROWS_ * DTK_;
            pack_dtlo_kernel<<<(int)((tot + 255) / 256), 256, 0, stream>>>(xdbl, dtlob);
        }

        // dt_proj: (8192x64) x (1536x64)^T -> dtbuf  [aligned, nk=1]
        gemm_fast(dtlob, dtwb + (size_t)i * DIN_ * DTK_, dtbuf, ROWS_, DIN_, DTK_);

        softplus_kernel<<<gUD, 256, 0, stream>>>(dtbuf, dtb + (size_t)i * DIN_);

        scan_kernel<<<B_ * (DIN_ / 256), 256, 0, stream>>>(
            uf, dtbuf, xdbl, xz,
            alog + (size_t)i * DIN_ * N_, dsk + (size_t)i * DIN_, yb);

        // out_proj: (8192x1536) x (768x1536)^T -> new h  [aligned, nk=24]
        gemm_fast(yb, opwb + (size_t)i * D_ * DIN_, hbuf, ROWS_, D_, DIN_);
    }

    final_head_kernel<<<B_, 256, 0, stream>>>(hbuf, resid, nfw, hw, hb, out);
}
